// ConvLayer_29472065585946
// MI455X (gfx1250) — matrix operand order repaired and checked by reading, not compiled, not run on hardware
//
#include <hip/hip_runtime.h>

// Problem constants (from reference): BS=2, N=4096, J=3, C=32, K=32
#define BSZ 2
#define NN  4096
#define JJ  3
#define CC  32
#define KK  32
#define RR  (NN * JJ)          // 12288 reduction length
#define SPLITS 4               // reduction split for occupancy / latency hiding
#define RSPLIT (RR / SPLITS)   // 3072
#define OUT_TOTAL (BSZ * KK * NN)  // 262144

typedef __attribute__((ext_vector_type(16))) _Float16 v16h;
typedef __attribute__((ext_vector_type(8)))  _Float16 h8;
typedef __attribute__((ext_vector_type(8)))  float    v8f;
typedef __attribute__((ext_vector_type(4)))  float    f4;

// ---------------------------------------------------------------------------
// Phase 1: T[b][k][m*J+j] = sum_c conv_w[k, j*C+c] * x[b,m,c]   (f16, 1.5 MB)
// One block per (b,m); 96 threads = (j,k) pairs; 32 FMAs each.
// ---------------------------------------------------------------------------
__global__ __launch_bounds__(96)
void precompute_T(const float* __restrict__ x, const float* __restrict__ cw,
                  _Float16* __restrict__ Tt) {
  const int bm = blockIdx.x;            // b*N + m
  const int b  = bm >> 12;              // / 4096
  const int m  = bm & (NN - 1);
  __shared__ float xs[CC];
  if (threadIdx.x < CC) xs[threadIdx.x] = x[(size_t)bm * CC + threadIdx.x];
  __syncthreads();
  const int k = threadIdx.x & 31;       // 0..31
  const int j = threadIdx.x >> 5;       // 0..2
  const float* wrow = cw + k * (JJ * CC) + j * CC;
  float acc = 0.f;
#pragma unroll
  for (int c = 0; c < CC; ++c) acc = fmaf(wrow[c], xs[c], acc);
  Tt[((size_t)(b * KK + k)) * RR + (size_t)m * JJ + j] = (_Float16)acc;
}

// ---------------------------------------------------------------------------
// Phase 2: WMMA GEMM. One wave per (split, batch, 16-row n-tile).
//   partial[s][b][k][n] = sum_{r in split s} W_row_n[r] * T[b][k][r]
// A tile: W rows (f32 -> f16 on the fly). B tile: Tt (already f16).
// Two v_wmma_f32_16x16x32_f16 per K-step cover all 32 output channels.
// ---------------------------------------------------------------------------
__global__ __launch_bounds__(32)
void gemm_wmma(const float* __restrict__ W, const _Float16* __restrict__ Tt,
               float* __restrict__ partial) {
  const int lane  = threadIdx.x;             // 0..31 (wave32)
  const int ntile = blockIdx.x & 255;        // N/16 = 256 tiles
  const int b     = (blockIdx.x >> 8) & 1;   // batch
  const int s     = blockIdx.x >> 9;         // split 0..3
  const int lhalf = lane >> 4;               // 0 or 1
  const int l16   = lane & 15;

  // A layout (16x32 f16): lane<16 -> row l16, K {0..7,16..23}; lane>=16 -> K {8..15,24..31}
  const int row = ntile * 16 + l16;
  const float* arow = W + ((size_t)(b * NN + row)) * RR;

  // B layout (32x16 f16): lanes 0-15 = col l16, K {0..7,16..23}; lanes 16-31 = K {8..15,24..31}
  const _Float16* t0 = Tt + ((size_t)(b * KK + l16)) * RR;       // k-cols 0..15
  const _Float16* t1 = Tt + ((size_t)(b * KK + 16 + l16)) * RR;  // k-cols 16..31

  v8f acc0 = {};
  v8f acc1 = {};
  int rr = s * RSPLIT;
  const int rend = rr + RSPLIT;

#pragma unroll 2
  for (; rr < rend; rr += 32) {
    const int ka = rr + (lhalf ? 8 : 0);
    // pull W ahead (~4KB) into the cache hierarchy
    __builtin_prefetch(arow + ka + 1024, 0, 1);
    f4 w0 = *(const f4*)(arow + ka);
    f4 w1 = *(const f4*)(arow + ka + 4);
    f4 w2 = *(const f4*)(arow + ka + 16);
    f4 w3 = *(const f4*)(arow + ka + 20);

    const int kb = rr + (lhalf ? 8 : 0);
    h8 b0lo = *(const h8*)(t0 + kb);
    h8 b0hi = *(const h8*)(t0 + kb + 16);
    h8 b1lo = *(const h8*)(t1 + kb);
    h8 b1hi = *(const h8*)(t1 + kb + 16);

    v16h a, bv0, bv1;
#pragma unroll
    for (int i = 0; i < 4; ++i) {
      a[i]      = (_Float16)w0[i];
      a[4 + i]  = (_Float16)w1[i];
      a[8 + i]  = (_Float16)w2[i];
      a[12 + i] = (_Float16)w3[i];
    }
#pragma unroll
    for (int i = 0; i < 8; ++i) {
      bv0[i] = b0lo[i]; bv0[8 + i] = b0hi[i];
      bv1[i] = b1lo[i]; bv1[8 + i] = b1hi[i];
    }

    acc0 = __builtin_amdgcn_wmma_f32_16x16x32_f16(false, a, false, bv0,
                                                  (short)0, acc0, false, false);
    acc1 = __builtin_amdgcn_wmma_f32_16x16x32_f16(false, a, false, bv1,
                                                  (short)0, acc1, false, false);
  }

  // C/D layout: VGPR r -> row M=r (lanes 0-15) / M=r+8 (lanes 16-31); col = lane%16.
  const int n0 = ntile * 16 + lhalf * 8;
  float* p0 = partial + ((size_t)((s * BSZ + b) * KK + l16)) * NN + n0;
  float* p1 = partial + ((size_t)((s * BSZ + b) * KK + 16 + l16)) * NN + n0;
  f4 lo0 = { acc0[0], acc0[1], acc0[2], acc0[3] };
  f4 hi0 = { acc0[4], acc0[5], acc0[6], acc0[7] };
  f4 lo1 = { acc1[0], acc1[1], acc1[2], acc1[3] };
  f4 hi1 = { acc1[4], acc1[5], acc1[6], acc1[7] };
  *(f4*)(p0)     = lo0;
  *(f4*)(p0 + 4) = hi0;
  *(f4*)(p1)     = lo1;
  *(f4*)(p1 + 4) = hi1;
}

// ---------------------------------------------------------------------------
// Phase 3: deterministic split reduction + bias.
// out[(b*K+k)*N+n] = sum_s partial[s][...] + conv_b[k]
// ---------------------------------------------------------------------------
__global__ __launch_bounds__(256)
void reduce_bias(const float* __restrict__ partial, const float* __restrict__ cb,
                 float* __restrict__ out) {
  const int i = blockIdx.x * 256 + threadIdx.x;  // i in [0, BS*K*N)
  float acc = cb[(i >> 12) & 31];                // (i / N) % K
#pragma unroll
  for (int p = 0; p < SPLITS; ++p)
    acc += partial[(size_t)p * OUT_TOTAL + i];
  out[i] = acc;
}

extern "C" void kernel_launch(void* const* d_in, const int* in_sizes, int n_in,
                              void* d_out, int out_size, void* d_ws, size_t ws_size,
                              hipStream_t stream) {
  const float* W  = (const float*)d_in[0];   // (BS, N, N, J) f32
  const float* x  = (const float*)d_in[1];   // (BS, N, C)    f32
  const float* cw = (const float*)d_in[2];   // (K, J*C)      f32
  const float* cb = (const float*)d_in[3];   // (K,)          f32
  float* out = (float*)d_out;                // (BS, K, N, 1) f32

  // Workspace layout: [ Tt : BS*K*R f16 = 1.5MB ][ partial : SPLITS*BS*K*N f32 = 4MB ]
  const size_t tt_bytes = (size_t)BSZ * KK * RR * sizeof(_Float16);  // 1,572,864 (256-aligned)
  _Float16* Tt   = (_Float16*)d_ws;
  float* partial = (float*)((char*)d_ws + tt_bytes);

  precompute_T<<<BSZ * NN, 96, 0, stream>>>(x, cw, Tt);
  gemm_wmma<<<BSZ * (NN / 16) * SPLITS, 32, 0, stream>>>(W, Tt, partial);
  reduce_bias<<<OUT_TOTAL / 256, 256, 0, stream>>>(partial, cb, out);
}
